// A2HNet_72464688218386
// MI455X (gfx1250) — matrix operand order repaired and checked by reading, not compiled, not run on hardware
//
#include <hip/hip_runtime.h>
#include <hip/hip_bf16.h>
#include <stdint.h>

// ---------------------------------------------------------------- types
typedef __attribute__((ext_vector_type(16))) _Float16 v16h;
typedef __attribute__((ext_vector_type(8)))  _Float16 v8h;
typedef __attribute__((ext_vector_type(2)))  _Float16 v2h;
typedef __attribute__((ext_vector_type(8)))  float    v8f;
typedef _Float16 half_t;

union F16x16 { v16h v; v8h h2[2]; };

#define N_NODES  102400
#define N_EDGES  409600
#define N_GRAPHS 2048
#define HID      128

// ---------------------------------------------------------------- async copy
// Per-lane async global->LDS transfer (VGPR-bypassing, tracked by ASYNCcnt).
__device__ __forceinline__ void async_load_b32(unsigned lds_off, const void* gaddr) {
    asm volatile("global_load_async_to_lds_b32 %0, %1, off"
                 :: "v"(lds_off), "v"((unsigned long long)(uintptr_t)gaddr)
                 : "memory");
}
__device__ __forceinline__ void async_wait0() {
    asm volatile("s_wait_asynccnt 0x0" ::: "memory");
}

// =================================================================
// Generic elementwise kernels
// =================================================================
__global__ void k_copy_f32(const float* __restrict__ in, float* __restrict__ out, int n) {
    int i = blockIdx.x * blockDim.x + threadIdx.x;
    if (i < n) out[i] = in[i];
}

__global__ void k_zero_f32(float* __restrict__ out, int n) {
    int i = blockIdx.x * blockDim.x + threadIdx.x;
    if (i < n) out[i] = 0.0f;
}

__global__ void k_cvt_f32_f16(const float* __restrict__ in, half_t* __restrict__ out, int n) {
    int i = blockIdx.x * blockDim.x + threadIdx.x;
    if (i < n) out[i] = (half_t)in[i];
}

// =================================================================
// Graph ops: scatter-add over edges, global add-pool (fp32 atomics)
// =================================================================
__global__ void k_scatter_add(const float* __restrict__ x, const int* __restrict__ ei,
                              float* __restrict__ agg, int K) {
    int e = blockIdx.x;
    int c = threadIdx.x;
    if (c >= K) return;
    int src = ei[e];
    int dst = ei[N_EDGES + e];
    atomicAdd(&agg[(size_t)dst * K + c], x[(size_t)src * K + c]);
}

__global__ void k_pool_add(const float* __restrict__ x, const int* __restrict__ batch,
                           float* __restrict__ xg) {
    int n = blockIdx.x;
    int c = threadIdx.x;   // 128 threads
    int g = batch[n];
    atomicAdd(&xg[(size_t)g * HID + c], x[(size_t)n * HID + c]);
}

// =================================================================
// BatchNorm over full node batch: per-column stats, then apply+ReLU
// =================================================================
__global__ void k_bn_stats(const float* __restrict__ h, float* __restrict__ mu,
                           float* __restrict__ rv, int M) {
    int c = blockIdx.x;   // column 0..127
    float s = 0.f, sq = 0.f;
    for (int r = threadIdx.x; r < M; r += blockDim.x) {
        float v = h[(size_t)r * HID + c];
        s += v; sq += v * v;
    }
    __shared__ float ls[256], lq[256];
    ls[threadIdx.x] = s; lq[threadIdx.x] = sq;
    __syncthreads();
    for (int off = 128; off > 0; off >>= 1) {
        if ((int)threadIdx.x < off) {
            ls[threadIdx.x] += ls[threadIdx.x + off];
            lq[threadIdx.x] += lq[threadIdx.x + off];
        }
        __syncthreads();
    }
    if (threadIdx.x == 0) {
        float mean = ls[0] / (float)M;
        float var  = lq[0] / (float)M - mean * mean;
        mu[c] = mean;
        rv[c] = rsqrtf(var + 1e-5f);
    }
}

__global__ void k_bn_apply(const float* __restrict__ h, const float* __restrict__ mu,
                           const float* __restrict__ rv, const float* __restrict__ g,
                           const float* __restrict__ b, float* __restrict__ xout, int n) {
    int i = blockIdx.x * blockDim.x + threadIdx.x;
    if (i >= n) return;
    int c = i & (HID - 1);
    float v = (h[i] - mu[c]) * rv[c] * g[c] + b[c];
    xout[i] = fmaxf(v, 0.0f);
}

// =================================================================
// WMMA GEMM: C[M x N] = act(A[M x K] @ B[K x N] + bias)
//   A, B f16 row-major.  Block = 128 threads (4 waves).
//   Wave w covers rows [blockIdx.x*64 + w*16, +16), cols [blockIdx.y*128, +128)
//   A tile (64x32) staged via GLOBAL_LOAD_ASYNC_TO_LDS_B32 (ASYNCcnt path);
//   B tile staged TRANSPOSED (BsT[n][kk], 128x32) so every lane fragment is
//   two aligned 16B chunks -> ds_load_b128 feeding v_wmma.
// =================================================================
template<bool RELU, bool OUT16>
__global__ __launch_bounds__(128)
void k_wmma_gemm(const half_t* __restrict__ A, const half_t* __restrict__ B,
                 const float* __restrict__ bias, void* __restrict__ Cout,
                 int M, int K, int Nb, int ldc, int col0) {
    __shared__ __align__(16) half_t As[64 * 32];     // [row][kk]
    __shared__ __align__(16) half_t BsT[128 * 32];   // [n][kk]

    const int tid  = threadIdx.x;
    const int wave = tid >> 5;
    const int lane = tid & 31;
    const int half = lane >> 4;     // which 16-lane half
    const int l15  = lane & 15;
    const int mrow0 = blockIdx.x * 64;
    const int ncol0 = blockIdx.y * 128;
    const unsigned as_base = (unsigned)(uintptr_t)(void*)As;  // LDS byte offset

    v8f acc[8];
#pragma unroll
    for (int i = 0; i < 8; ++i)
#pragma unroll
        for (int j = 0; j < 8; ++j) acc[i][j] = 0.0f;

    for (int k0 = 0; k0 < K; k0 += 32) {
        // prefetch next A tile (one cacheline probe per thread)
        if (k0 + 32 < K) {
            int prow = mrow0 + (tid & 63);
            if (prow < M)
                __builtin_prefetch(A + (size_t)prow * K + k0 + 32, 0, 1);
        }
        // ---- stage A tile: 64 rows x 32 kk, async global->LDS b32 pairs ----
#pragma unroll
        for (int i = tid; i < 64 * 16; i += 128) {
            int r = i >> 4, kk = (i & 15) * 2;
            int grow = mrow0 + r, gk = k0 + kk;
            unsigned loff = as_base + (unsigned)(r * 32 + kk) * 2u;
            if (grow < M && gk + 1 < K) {
                async_load_b32(loff, A + (size_t)grow * K + gk);   // no VGPR return
            } else {
                v2h p;
                p.x = (grow < M && gk < K) ? A[(size_t)grow * K + gk] : (half_t)0;
                p.y = (half_t)0;
                *(v2h*)(As + r * 32 + kk) = p;                     // ds_store_b32
            }
        }
        // ---- stage B tile transposed: BsT[n][kk], b32 pairs along kk ----
#pragma unroll
        for (int i = tid; i < 128 * 16; i += 128) {
            int nn = i & 127, kk = (i >> 7) * 2;
            int gk = k0 + kk;
            v2h p;
            p.x = (gk     < K) ? B[(size_t)gk * Nb + ncol0 + nn]       : (half_t)0;
            p.y = (gk + 1 < K) ? B[(size_t)(gk + 1) * Nb + ncol0 + nn] : (half_t)0;
            *(v2h*)(BsT + nn * 32 + kk) = p;                      // ds_store_b32
        }
        async_wait0();        // drain this wave's async LDS writes
        __syncthreads();

        // ---- A fragment: two contiguous 16B chunks -> 2x ds_load_b128 ----
        const int rl = wave * 16 + l15;
        F16x16 af;
        af.h2[0] = *(const v8h*)(As + rl * 32 + half * 8);
        af.h2[1] = *(const v8h*)(As + rl * 32 + 16 + half * 8);

#pragma unroll
        for (int nt = 0; nt < 8; ++nt) {
            const half_t* bp = BsT + (nt * 16 + l15) * 32 + half * 8;
            F16x16 bf;
            bf.h2[0] = *(const v8h*)(bp);          // ds_load_b128
            bf.h2[1] = *(const v8h*)(bp + 16);     // ds_load_b128
            acc[nt] = __builtin_amdgcn_wmma_f32_16x16x32_f16(
                false, af.v, false, bf.v, (short)0, acc[nt], false, false);
        }
        __syncthreads();
    }

    // epilogue: C/D 16x16 f32 layout (8 VGPRs, lanes split by half)
    const int rbase = mrow0 + wave * 16 + half * 8;
#pragma unroll
    for (int nt = 0; nt < 8; ++nt) {
        int ncol = ncol0 + nt * 16 + l15;
        float bv = bias ? bias[ncol] : 0.0f;
#pragma unroll
        for (int r = 0; r < 8; ++r) {
            int row = rbase + r;
            if (row < M) {
                float v = acc[nt][r] + bv;
                if (RELU) v = fmaxf(v, 0.0f);
                size_t cidx = (size_t)row * ldc + col0 + ncol;
                if (OUT16) ((half_t*)Cout)[cidx] = (half_t)v;
                else       ((float*)Cout)[cidx]  = v;
            }
        }
    }
}

// =================================================================
// Protein branch: Conv1d(1->32, k=8) + ReLU + global max pool -> f16
// one block (256 threads = 8 waves) per graph
// =================================================================
__global__ __launch_bounds__(256)
void k_conv_prot(const float* __restrict__ seq, const float* __restrict__ ck,
                 const float* __restrict__ cb, half_t* __restrict__ xph) {
    int g = blockIdx.x;
    __shared__ float s_seq[1000];
    __shared__ float s_k[32 * 8];
    __shared__ float s_b[32];
    __shared__ float s_red[8 * 32];
    for (int i = threadIdx.x; i < 1000; i += 256) s_seq[i] = seq[(size_t)g * 1000 + i];
    if (threadIdx.x < 256) s_k[threadIdx.x] = ck[threadIdx.x];
    if (threadIdx.x < 32)  s_b[threadIdx.x] = cb[threadIdx.x];
    __syncthreads();

    float mx[32];
#pragma unroll
    for (int o = 0; o < 32; ++o) mx[o] = 0.0f;   // max of relu >= 0

    for (int t = threadIdx.x; t < 993; t += 256) {
#pragma unroll
        for (int o = 0; o < 32; ++o) {
            float acc = s_b[o];
#pragma unroll
            for (int k = 0; k < 8; ++k) acc = fmaf(s_seq[t + k], s_k[o * 8 + k], acc);
            mx[o] = fmaxf(mx[o], fmaxf(acc, 0.0f));
        }
    }
    int lane = threadIdx.x & 31, wv = threadIdx.x >> 5;
#pragma unroll
    for (int o = 0; o < 32; ++o)
        for (int off = 16; off > 0; off >>= 1)
            mx[o] = fmaxf(mx[o], __shfl_down(mx[o], off, 32));
    if (lane == 0)
#pragma unroll
        for (int o = 0; o < 32; ++o) s_red[wv * 32 + o] = mx[o];
    __syncthreads();
    if (threadIdx.x < 32) {
        float m = s_red[threadIdx.x];
#pragma unroll
        for (int w = 1; w < 8; ++w) m = fmaxf(m, s_red[w * 32 + threadIdx.x]);
        xph[(size_t)g * 32 + threadIdx.x] = (half_t)m;
    }
}

// =================================================================
// Final head: out[g] = xc2[g,:] . out_W + out_b
// =================================================================
__global__ void k_final(const half_t* __restrict__ xc2, const float* __restrict__ W,
                        const float* __restrict__ b, float* __restrict__ out) {
    int g = blockIdx.x * blockDim.x + threadIdx.x;
    if (g >= N_GRAPHS) return;
    float acc = b[0];
#pragma unroll 4
    for (int c = 0; c < HID; ++c) acc = fmaf((float)xc2[(size_t)g * HID + c], W[c], acc);
    out[g] = acc;
}

// =================================================================
// Host driver
// =================================================================
static inline int cdiv(int a, int b) { return (a + b - 1) / b; }

extern "C" void kernel_launch(void* const* d_in, const int* in_sizes, int n_in,
                              void* d_out, int out_size, void* d_ws, size_t ws_size,
                              hipStream_t stream) {
    // -------- inputs (setup_inputs order) --------
    const float* x_ligand    = (const float*)d_in[0];   // 102400 x 78
    const float* protein_seq = (const float*)d_in[1];   // 2048 x 1000
    const float* a2h         = (const float*)d_in[2];   // 2048 x 3000
    const int*   edge_index  = (const int*)d_in[3];     // 2 x 409600
    const int*   batch_lig   = (const int*)d_in[4];     // 102400
    const float* gW1[3] = { (const float*)d_in[5],  (const float*)d_in[11], (const float*)d_in[17] };
    const float* gb1[3] = { (const float*)d_in[6],  (const float*)d_in[12], (const float*)d_in[18] };
    const float* gW2[3] = { (const float*)d_in[7],  (const float*)d_in[13], (const float*)d_in[19] };
    const float* gb2[3] = { (const float*)d_in[8],  (const float*)d_in[14], (const float*)d_in[20] };
    const float* bng[3] = { (const float*)d_in[9],  (const float*)d_in[15], (const float*)d_in[21] };
    const float* bnb[3] = { (const float*)d_in[10], (const float*)d_in[16], (const float*)d_in[22] };
    const float* lig_W  = (const float*)d_in[23];  const float* lig_b  = (const float*)d_in[24];
    const float* conv_k = (const float*)d_in[25];  const float* conv_b = (const float*)d_in[26];
    const float* prot_W = (const float*)d_in[27];  const float* prot_b = (const float*)d_in[28];
    const float* a2h1_W = (const float*)d_in[29];  const float* a2h1_b = (const float*)d_in[30];
    const float* a2h2_W = (const float*)d_in[31];  const float* a2h2_b = (const float*)d_in[32];
    const float* c1_W   = (const float*)d_in[33];  const float* c1_b   = (const float*)d_in[34];
    const float* c2_W   = (const float*)d_in[35];  const float* c2_b   = (const float*)d_in[36];
    const float* out_W  = (const float*)d_in[37];  const float* out_b  = (const float*)d_in[38];
    float* out = (float*)d_out;

    // -------- workspace layout --------
    char* ws = (char*)d_ws;
    size_t off = 0;
    auto alloc = [&](size_t bytes) { size_t r = off; off += (bytes + 255) & ~(size_t)255; return r; };
    const size_t O_XBUF = alloc((size_t)N_NODES * HID * 4);   // current x (f32)
    const size_t O_AGG  = alloc((size_t)N_NODES * HID * 4);   // agg / h (f32); reused post-GIN
    const size_t O_XH16 = alloc((size_t)N_NODES * HID * 2);   // f16 GEMM A
    const size_t O_T16  = alloc((size_t)N_NODES * HID * 2);   // f16 MLP hidden
    const size_t O_W16  = alloc((size_t)3000 * HID * 2);      // f16 weight staging (largest: a2h1_W)
    const size_t O_MU   = alloc(HID * 4);
    const size_t O_RV   = alloc(HID * 4);
    // post-GIN branch buffers overlaid into the freed AGG region
    size_t boff = O_AGG;
    auto balloc = [&](size_t bytes) { size_t r = boff; boff += (bytes + 255) & ~(size_t)255; return r; };
    const size_t O_XA16   = balloc((size_t)N_GRAPHS * 3000 * 2);
    const size_t O_XG     = balloc((size_t)N_GRAPHS * HID * 4);
    const size_t O_XG16   = balloc((size_t)N_GRAPHS * HID * 2);
    const size_t O_XAM16  = balloc((size_t)N_GRAPHS * HID * 2);
    const size_t O_XPH16  = balloc((size_t)N_GRAPHS * 32 * 2);
    const size_t O_XC16   = balloc((size_t)N_GRAPHS * 384 * 2);
    const size_t O_C1O16  = balloc((size_t)N_GRAPHS * 256 * 2);
    const size_t O_C2O16  = balloc((size_t)N_GRAPHS * HID * 2);

    float*  XBUF = (float*)(ws + O_XBUF);
    float*  AGG  = (float*)(ws + O_AGG);
    half_t* XH16 = (half_t*)(ws + O_XH16);
    half_t* T16  = (half_t*)(ws + O_T16);
    half_t* W16  = (half_t*)(ws + O_W16);
    float*  MU   = (float*)(ws + O_MU);
    float*  RV   = (float*)(ws + O_RV);
    half_t* XA16  = (half_t*)(ws + O_XA16);
    float*  XG    = (float*)(ws + O_XG);
    half_t* XG16  = (half_t*)(ws + O_XG16);
    half_t* XAM16 = (half_t*)(ws + O_XAM16);
    half_t* XPH16 = (half_t*)(ws + O_XPH16);
    half_t* XC16  = (half_t*)(ws + O_XC16);
    half_t* C1O16 = (half_t*)(ws + O_C1O16);
    half_t* C2O16 = (half_t*)(ws + O_C2O16);

    const int TB = 256;
    auto cvt = [&](const float* src, half_t* dst, int n) {
        k_cvt_f32_f16<<<cdiv(n, TB), TB, 0, stream>>>(src, dst, n);
    };

    // ================= GIN layers =================
    for (int L = 0; L < 3; ++L) {
        const float* xin = (L == 0) ? x_ligand : XBUF;
        const int Kin = (L == 0) ? 78 : HID;
        const int nelem = N_NODES * Kin;
        // h_pre = x + segment_sum(x[src] -> dst)
        k_copy_f32<<<cdiv(nelem, TB), TB, 0, stream>>>(xin, AGG, nelem);
        k_scatter_add<<<N_EDGES, 128, 0, stream>>>(xin, edge_index, AGG, Kin);
        cvt(AGG, XH16, nelem);
        // t = relu(h_pre @ W1 + b1)    (f16 out)
        cvt(gW1[L], W16, Kin * HID);
        k_wmma_gemm<true, true><<<dim3(N_NODES / 64, 1), 128, 0, stream>>>(
            XH16, W16, gb1[L], (void*)T16, N_NODES, Kin, HID, HID, 0);
        // h = t @ W2 + b2              (f32 out, into AGG)
        cvt(gW2[L], W16, HID * HID);
        k_wmma_gemm<false, false><<<dim3(N_NODES / 64, 1), 128, 0, stream>>>(
            T16, W16, gb2[L], (void*)AGG, N_NODES, HID, HID, HID, 0);
        // batch-norm (batch stats) + relu -> XBUF
        k_bn_stats<<<HID, 256, 0, stream>>>(AGG, MU, RV, N_NODES);
        k_bn_apply<<<cdiv(N_NODES * HID, TB), TB, 0, stream>>>(
            AGG, MU, RV, bng[L], bnb[L], XBUF, N_NODES * HID);
    }

    // ================= ligand pooling branch =================
    k_zero_f32<<<cdiv(N_GRAPHS * HID, TB), TB, 0, stream>>>(XG, N_GRAPHS * HID);
    k_pool_add<<<N_NODES, HID, 0, stream>>>(XBUF, batch_lig, XG);
    cvt(XG, XG16, N_GRAPHS * HID);
    cvt(lig_W, W16, HID * HID);
    k_wmma_gemm<true, true><<<dim3(N_GRAPHS / 64, 1), 128, 0, stream>>>(
        XG16, W16, lig_b, (void*)XC16, N_GRAPHS, HID, HID, 384, 0);

    // ================= protein CNN branch =================
    k_conv_prot<<<N_GRAPHS, 256, 0, stream>>>(protein_seq, conv_k, conv_b, XPH16);
    cvt(prot_W, W16, 32 * HID);
    k_wmma_gemm<true, true><<<dim3(N_GRAPHS / 64, 1), 128, 0, stream>>>(
        XPH16, W16, prot_b, (void*)XC16, N_GRAPHS, 32, HID, 384, HID);

    // ================= a2h MLP branch =================
    cvt(a2h, XA16, N_GRAPHS * 3000);
    cvt(a2h1_W, W16, 3000 * HID);
    k_wmma_gemm<true, true><<<dim3(N_GRAPHS / 64, 1), 128, 0, stream>>>(
        XA16, W16, a2h1_b, (void*)XAM16, N_GRAPHS, 3000, HID, HID, 0);
    cvt(a2h2_W, W16, HID * HID);
    k_wmma_gemm<true, true><<<dim3(N_GRAPHS / 64, 1), 128, 0, stream>>>(
        XAM16, W16, a2h2_b, (void*)XC16, N_GRAPHS, HID, HID, 384, 256);

    // ================= fused head =================
    cvt(c1_W, W16, 384 * 256);
    k_wmma_gemm<true, true><<<dim3(N_GRAPHS / 64, 2), 128, 0, stream>>>(
        XC16, W16, c1_b, (void*)C1O16, N_GRAPHS, 384, 256, 256, 0);
    cvt(c2_W, W16, 256 * HID);
    k_wmma_gemm<true, true><<<dim3(N_GRAPHS / 64, 1), 128, 0, stream>>>(
        C1O16, W16, c2_b, (void*)C2O16, N_GRAPHS, 256, HID, HID, 0);
    k_final<<<cdiv(N_GRAPHS, TB), TB, 0, stream>>>(C2O16, out_W, out_b, out);

    (void)in_sizes; (void)n_in; (void)out_size; (void)ws_size;
}
